// SingleHeadAttention_9861244912115
// MI455X (gfx1250) — matrix operand order repaired and checked by reading, not compiled, not run on hardware
//
#include <hip/hip_runtime.h>
#include <hip/hip_bf16.h>

#define BATCH 4
#define SEQ   2048
#define DIM   512
#define TDIM  1536   // 3*DIM

typedef _Float16 v16h __attribute__((ext_vector_type(16)));
typedef _Float16 v8h  __attribute__((ext_vector_type(8)));
typedef float    v8f  __attribute__((ext_vector_type(8)));
typedef float    v4f  __attribute__((ext_vector_type(4)));

static __device__ __forceinline__ v16h cat16(v8h lo, v8h hi) {
  return __builtin_shufflevector(lo, hi, 0,1,2,3,4,5,6,7,8,9,10,11,12,13,14,15);
}

static __device__ __forceinline__ v8f wmma_f16(v16h a, v16h b, v8f c) {
  return __builtin_amdgcn_wmma_f32_16x16x32_f16(false, a, false, b, (short)0, c,
                                                false, false);
}

// Async cache->LDS b128 copy (CDNA5 GLOBAL_LOAD_ASYNC_TO_LDS_B128, ASYNCcnt).
// lds_off: wave-relative LDS byte offset (low 32 bits of a generic pointer to
// a __shared__ object), gaddr: 64-bit global address.
static __device__ __forceinline__ void async_g2l_b128(unsigned lds_off,
                                                      const void* gaddr) {
  asm volatile("global_load_async_to_lds_b128 %0, %1, off"
               :: "v"(lds_off), "v"(gaddr)
               : "memory");
}
static __device__ __forceinline__ void wait_async0() {
  asm volatile("s_wait_asynccnt 0" ::: "memory");
}

// Stage `bytes16` x 16B from gsrc into LDS at lds_base, 256 threads.
static __device__ __forceinline__ void stage_tile_async(
    unsigned lds_base, const void* gsrc, int n16, int tid) {
  const char* g = (const char*)gsrc;
  for (int i = tid; i < n16; i += 256) {
    async_g2l_b128(lds_base + (unsigned)i * 16u, g + (size_t)i * 16u);
  }
  wait_async0();
}

// ---------------------------------------------------------------------------
// Kernel 0: one-time fp32 -> f16 conversion (x, qkv_w, out_w).
// ---------------------------------------------------------------------------
__global__ __launch_bounds__(256) void cvt_f16_kernel(
    const float* __restrict__ in, _Float16* __restrict__ out) {
  const size_t i = ((size_t)blockIdx.x * 256 + threadIdx.x) * 8;
  v4f a = *(const v4f*)(in + i);
  v4f b = *(const v4f*)(in + i + 4);
  v8h r;
  #pragma unroll
  for (int j = 0; j < 4; ++j) {
    r[j]     = (_Float16)a[j];
    r[4 + j] = (_Float16)b[j];
  }
  *(v8h*)(out + i) = r;
}

// ---------------------------------------------------------------------------
// Kernel 1: QKV projection (all-f16 operands, f32 accumulate).
// Block = 8 waves; 16x512 A tile async-staged to LDS; 4 n-tiles per wave.
// blockIdx.y in {0,1,2} selects the Q / K / V third of the 1536 columns.
// Q scaled by D^-0.5; V stored transposed [B, D, S].
// ---------------------------------------------------------------------------
__global__ __launch_bounds__(256) void qkv_kernel(
    const _Float16* __restrict__ Xh, const _Float16* __restrict__ Wh,
    const float* __restrict__ bias,
    _Float16* __restrict__ Q, _Float16* __restrict__ Kc,
    _Float16* __restrict__ Vt) {
  __shared__ __align__(16) _Float16 atile[16 * DIM];   // 16 KB

  const int tid  = threadIdx.x;
  const int lane = tid & 31;
  const int wv   = tid >> 5;
  const int m0   = blockIdx.x * 16;
  const int ntb  = blockIdx.y * 32 + wv * 4;   // first of 4 n-tiles (0..95)

  stage_tile_async((unsigned)(uintptr_t)atile, Xh + (size_t)m0 * DIM,
                   16 * DIM * 2 / 16, tid);
  __syncthreads();

  const int kh    = (lane >> 4) << 3;   // A K-half {0,8}
  const int kbh   = (lane >> 4) << 3;   // B K-half {0,8}
  const int am    = lane & 15;
  const int rbase = (lane >> 4) << 3;

  const _Float16* bp[4];
  #pragma unroll
  for (int j = 0; j < 4; ++j)
    bp[j] = Wh + ((size_t)(ntb + j) * 16 + am) * DIM + kbh;

  v8f acc[4];
  #pragma unroll
  for (int j = 0; j < 4; ++j) acc[j] = (v8f){0, 0, 0, 0, 0, 0, 0, 0};

  #pragma unroll 2
  for (int kc = 0; kc < DIM; kc += 32) {
    __builtin_prefetch(bp[0] + kc + 128, 0, 1);
    const _Float16* ap = atile + am * DIM + kc + kh;
    v16h a = cat16(*(const v8h*)ap, *(const v8h*)(ap + 16));
    #pragma unroll
    for (int j = 0; j < 4; ++j) {
      v16h b = cat16(*(const v8h*)(bp[j] + kc), *(const v8h*)(bp[j] + kc + 16));
      acc[j] = wmma_f16(a, b, acc[j]);
    }
  }

  const float scale = 0.04419417382415922f;  // 512^-0.5
  #pragma unroll
  for (int j = 0; j < 4; ++j) {
    const int bn = (ntb + j) * 16 + am;      // global embed column (0..1535)
    const float bv = bias[bn];
    #pragma unroll
    for (int i = 0; i < 8; ++i) {
      const int row = m0 + rbase + i;        // global token index
      float v = acc[j][i] + bv;
      if (bn < DIM) {
        Q[(size_t)row * DIM + bn] = (_Float16)(v * scale);
      } else if (bn < 2 * DIM) {
        Kc[(size_t)row * DIM + (bn - DIM)] = (_Float16)v;
      } else {
        const int bb = row >> 11;
        const int s  = row & (SEQ - 1);
        Vt[((size_t)bb * DIM + (bn - 2 * DIM)) * SEQ + s] = (_Float16)v;
      }
    }
  }
}

// ---------------------------------------------------------------------------
// Kernel 2: flash attention.  One block (8 waves) per 16-query tile.
// Wave w computes one 16x16 score tile per 128-key block, online softmax
// state in LDS, then accumulates its 64-wide D slice of P*V in VGPRs.
// ---------------------------------------------------------------------------
__global__ __launch_bounds__(256) void attn_kernel(
    const _Float16* __restrict__ Q, const _Float16* __restrict__ Kc,
    const _Float16* __restrict__ Vt, _Float16* __restrict__ AO) {
  __shared__ __align__(16) _Float16 qtile[16 * DIM];   // 16 KB
  __shared__ __align__(16) float    sblk[16 * 128];    // 8 KB
  __shared__ __align__(16) _Float16 pblk[16 * 128];    // 4 KB
  __shared__ float rowm[16], rowl[16], rowc[16];

  const int tid  = threadIdx.x;
  const int lane = tid & 31;
  const int wv   = tid >> 5;
  const int b    = blockIdx.y;
  const int q0   = blockIdx.x * 16;

  stage_tile_async((unsigned)(uintptr_t)qtile,
                   Q + ((size_t)b * SEQ + q0) * DIM, 16 * DIM * 2 / 16, tid);
  if (tid < 16) { rowm[tid] = -3.0e38f; rowl[tid] = 0.0f; }

  v8f Oacc[4];
  #pragma unroll
  for (int nt = 0; nt < 4; ++nt) Oacc[nt] = (v8f){0, 0, 0, 0, 0, 0, 0, 0};

  const int kh    = (lane >> 4) << 3;
  const int kbh   = (lane >> 4) << 3;
  const int rbase = kh;
  const int am    = lane & 15;

  __syncthreads();

  for (int jb = 0; jb < SEQ; jb += 128) {
    // ---- scores: this wave's 16 keys vs the shared 16 queries ----
    const _Float16* kp =
        Kc + ((size_t)b * SEQ + jb + wv * 16 + (lane & 15)) * DIM;
    v8f sfrag = {0, 0, 0, 0, 0, 0, 0, 0};
    #pragma unroll 4
    for (int kc = 0; kc < DIM; kc += 32) {
      const _Float16* qp = qtile + am * DIM + kc + kh;
      v16h a = cat16(*(const v8h*)qp, *(const v8h*)(qp + 16));
      const _Float16* bpk = kp + kc + kbh;
      v16h bm = cat16(*(const v8h*)bpk, *(const v8h*)(bpk + 16));
      sfrag = wmma_f16(a, bm, sfrag);
    }
    {
      const int col = wv * 16 + (lane & 15);
      #pragma unroll
      for (int i = 0; i < 8; ++i) sblk[(rbase + i) * 128 + col] = sfrag[i];
    }
    __syncthreads();

    // ---- online softmax state update (one thread per query row) ----
    if (tid < 16) {
      const float* sr = sblk + tid * 128;
      float bmax = -3.0e38f;
      for (int j = 0; j < 128; ++j) bmax = fmaxf(bmax, sr[j]);
      const float mo = rowm[tid];
      const float mn = fmaxf(mo, bmax);
      rowc[tid] = __expf(mo - mn);
      rowm[tid] = mn;
    }
    __syncthreads();

    for (int idx = tid; idx < 16 * 128; idx += 256) {
      const int r = idx >> 7;
      pblk[idx] = (_Float16)__expf(sblk[idx] - rowm[r]);
    }
    __syncthreads();

    if (tid < 16) {
      const _Float16* pr = pblk + tid * 128;
      float s = 0.0f;
      for (int j = 0; j < 128; ++j) s += (float)pr[j];
      rowl[tid] = rowl[tid] * rowc[tid] + s;
    }

    // ---- rescale accumulators, then P * V for this wave's D slice ----
    float cf[8];
    #pragma unroll
    for (int i = 0; i < 8; ++i) cf[i] = rowc[rbase + i];
    #pragma unroll
    for (int nt = 0; nt < 4; ++nt)
      #pragma unroll
      for (int i = 0; i < 8; ++i) Oacc[nt][i] *= cf[i];

    for (int kc2 = 0; kc2 < 128; kc2 += 32) {
      const _Float16* pp = pblk + am * 128 + kc2 + kh;
      v16h a = cat16(*(const v8h*)pp, *(const v8h*)(pp + 16));
      #pragma unroll
      for (int nt = 0; nt < 4; ++nt) {
        const int d = wv * 64 + nt * 16 + (lane & 15);
        const _Float16* vp =
            Vt + ((size_t)b * DIM + d) * SEQ + jb + kc2 + kbh;
        v16h bm = cat16(*(const v8h*)vp, *(const v8h*)(vp + 16));
        Oacc[nt] = wmma_f16(a, bm, Oacc[nt]);
      }
    }
    __syncthreads();
  }

  float li[8];
  #pragma unroll
  for (int i = 0; i < 8; ++i) li[i] = 1.0f / rowl[rbase + i];
  #pragma unroll
  for (int nt = 0; nt < 4; ++nt) {
    const int d = wv * 64 + nt * 16 + (lane & 15);
    #pragma unroll
    for (int i = 0; i < 8; ++i) {
      const int row = q0 + rbase + i;
      AO[((size_t)b * SEQ + row) * DIM + d] = (_Float16)(Oacc[nt][i] * li[i]);
    }
  }
}

// ---------------------------------------------------------------------------
// Kernel 3: output projection (all-f16 operands, f32 accumulate + bias).
// ---------------------------------------------------------------------------
__global__ __launch_bounds__(256) void proj_kernel(
    const _Float16* __restrict__ AO, const _Float16* __restrict__ Wh,
    const float* __restrict__ bias, float* __restrict__ out) {
  __shared__ __align__(16) _Float16 atile[16 * DIM];   // 16 KB

  const int tid  = threadIdx.x;
  const int lane = tid & 31;
  const int wv   = tid >> 5;
  const int m0   = blockIdx.x * 16;
  const int ntb  = wv * 4;                      // 4 n-tiles per wave, 32 total

  stage_tile_async((unsigned)(uintptr_t)atile, AO + (size_t)m0 * DIM,
                   16 * DIM * 2 / 16, tid);
  __syncthreads();

  const int kh    = (lane >> 4) << 3;
  const int kbh   = (lane >> 4) << 3;
  const int am    = lane & 15;
  const int rbase = (lane >> 4) << 3;

  const _Float16* bp[4];
  #pragma unroll
  for (int j = 0; j < 4; ++j)
    bp[j] = Wh + ((size_t)(ntb + j) * 16 + am) * DIM + kbh;

  v8f acc[4];
  #pragma unroll
  for (int j = 0; j < 4; ++j) acc[j] = (v8f){0, 0, 0, 0, 0, 0, 0, 0};

  #pragma unroll 2
  for (int kc = 0; kc < DIM; kc += 32) {
    __builtin_prefetch(bp[0] + kc + 128, 0, 1);
    const _Float16* ap = atile + am * DIM + kc + kh;
    v16h a = cat16(*(const v8h*)ap, *(const v8h*)(ap + 16));
    #pragma unroll
    for (int j = 0; j < 4; ++j) {
      v16h b = cat16(*(const v8h*)(bp[j] + kc), *(const v8h*)(bp[j] + kc + 16));
      acc[j] = wmma_f16(a, b, acc[j]);
    }
  }

  #pragma unroll
  for (int j = 0; j < 4; ++j) {
    const int bn = (ntb + j) * 16 + am;
    const float bv = bias[bn];
    #pragma unroll
    for (int i = 0; i < 8; ++i) {
      out[(size_t)(m0 + rbase + i) * DIM + bn] = acc[j][i] + bv;
    }
  }
}

// ---------------------------------------------------------------------------
extern "C" void kernel_launch(void* const* d_in, const int* in_sizes, int n_in,
                              void* d_out, int out_size, void* d_ws,
                              size_t ws_size, hipStream_t stream) {
  const float* x     = (const float*)d_in[0];
  const float* qkv_w = (const float*)d_in[1];
  const float* qkv_b = (const float*)d_in[2];
  const float* out_w = (const float*)d_in[3];
  const float* out_b = (const float*)d_in[4];
  float* out = (float*)d_out;

  const size_t NX  = (size_t)BATCH * SEQ * DIM;  // 4,194,304
  const size_t NWQ = (size_t)TDIM * DIM;         //   786,432
  const size_t NWO = (size_t)DIM * DIM;          //   262,144

  _Float16* Xh  = (_Float16*)d_ws;
  _Float16* Wqh = Xh + NX;
  _Float16* Woh = Wqh + NWQ;
  _Float16* Q   = Woh + NWO;
  _Float16* Kc  = Q + NX;
  _Float16* Vt  = Kc + NX;   // transposed V: [B, D, S]
  _Float16* AO  = Vt + NX;   // pre-projection attention output

  cvt_f16_kernel<<<dim3(NX / 2048), 256, 0, stream>>>(x, Xh);
  cvt_f16_kernel<<<dim3(NWQ / 2048), 256, 0, stream>>>(qkv_w, Wqh);
  cvt_f16_kernel<<<dim3(NWO / 2048), 256, 0, stream>>>(out_w, Woh);

  qkv_kernel<<<dim3(512, 3), 256, 0, stream>>>(Xh, Wqh, qkv_b, Q, Kc, Vt);
  attn_kernel<<<dim3(SEQ / 16, BATCH), 256, 0, stream>>>(Q, Kc, Vt, AO);
  proj_kernel<<<dim3(512, 1), 256, 0, stream>>>(AO, Woh, out_b, out);
}
